// my_model_25074019074361
// MI455X (gfx1250) — matrix operand order repaired and checked by reading, not compiled, not run on hardware
//
#include <hip/hip_runtime.h>
#include <hip/hip_bf16.h>

// ---------------------------------------------------------------------------
// Problem constants
// ---------------------------------------------------------------------------
#define B_     4
#define L_     1024
#define H_     768
#define HEADS_ 12
#define M_     60
#define E_     30
#define P_     870
#define R_     97
#define BLK_   64
#define NB_    12                 // H_/BLK_
#define PPAD   880                // P_ padded to multiple of 16
#define PBPAD  (PPAD * B_)        // 3520 GEMM rows
#define RPAD   112                // R_ padded to multiple of 16
#define KCLS   (H_ * BLK_)        // 49152 classifier K
#define EPS_   1e-20f

typedef __attribute__((ext_vector_type(16))) __bf16 v16bf;
typedef __attribute__((ext_vector_type(8)))  float  v8f;
typedef __attribute__((ext_vector_type(4)))  unsigned int u32x4;

// ---------------------------------------------------------------------------
// WMMA fragment helpers (bf16 A 16x32 / B 32x16 / C f32 16x16, wave32)
// A (row-major M x K): lane = {row = lane&15, half = lane>>4};
//   V0-3 hold K = k+8h..k+8h+7, V4-7 hold K = k+16+8h..+7  -> two 16B loads.
// B as W (row-major N x K): lane holds col n = lane&15, V0-3 K = k+8h..+7,
//   V4-7 K = k+16+8h..+7 -> two 16B loads from row n.
// C/D: lane l, elem v -> row = v + 8*(l>>4), col = l&15.
// ---------------------------------------------------------------------------
__device__ __forceinline__ v16bf frag_ld2(const __bf16* p0, const __bf16* p1) {
  union { v16bf v; u32x4 u[2]; } r;
  r.u[0] = *reinterpret_cast<const u32x4*>(p0);
  r.u[1] = *reinterpret_cast<const u32x4*>(p1);
  return r.v;
}

__device__ __forceinline__ v16bf load_frag_a(const __bf16* base, long lda, int row0, long k) {
  int lane = threadIdx.x & 31;
  int half = lane >> 4;
  const __bf16* r = base + (long)(row0 + (lane & 15)) * lda + k + 8 * half;
  return frag_ld2(r, r + 16);
}

__device__ __forceinline__ v16bf load_frag_b(const __bf16* base, long ldb, int col0, long k) {
  int lane = threadIdx.x & 31;
  int half = lane >> 4;
  const __bf16* r = base + (long)(col0 + (lane & 15)) * ldb + k + 8 * half;
  return frag_ld2(r, r + 16);
}

__device__ __forceinline__ v8f wmma_bf16(v16bf a, v16bf b, v8f c) {
  return __builtin_amdgcn_wmma_f32_16x16x32_bf16(false, a, false, b, (short)0, c,
                                                 false, false);
}

__device__ __forceinline__ void store_frag_bf16(__bf16* base, long ldc, int row0,
                                                int col0, v8f c) {
  int lane = threadIdx.x & 31;
  int col = col0 + (lane & 15);
  int rbase = row0 + 8 * (lane >> 4);
#pragma unroll
  for (int v = 0; v < 8; ++v)
    base[(long)(rbase + v) * ldc + col] = (__bf16)c[v];
}

__device__ __forceinline__ v16bf scale_frag(v16bf t, float s) {
  v16bf r;
#pragma unroll
  for (int q = 0; q < 16; ++q) r[q] = (__bf16)(s * (float)t[q]);
  return r;
}

// ---------------------------------------------------------------------------
// Stage 1: entity[b,e,h] = log( sum_m em[b,e,m] * exp(context[b, mm[b,m], h]) )
// ---------------------------------------------------------------------------
__global__ void entity_kernel(const float* __restrict__ ctx,
                              const float* __restrict__ em,
                              const int*   __restrict__ mm,
                              float* __restrict__ entity) {
  int b = blockIdx.x / E_, e = blockIdx.x % E_;
  __shared__ float em_s[M_];
  __shared__ int   mm_s[M_];
  if (threadIdx.x < M_) {
    em_s[threadIdx.x] = em[((long)b * E_ + e) * M_ + threadIdx.x];
    mm_s[threadIdx.x] = mm[(long)b * M_ + threadIdx.x];
  }
  __syncthreads();
  for (int h = threadIdx.x; h < H_; h += blockDim.x) {
    float acc = 0.f;
    for (int m = 0; m < M_; ++m)
      acc += em_s[m] * expf(ctx[((long)b * L_ + mm_s[m]) * H_ + h]);
    entity[((long)b * E_ + e) * H_ + h] = logf(acc);
  }
}

// ---------------------------------------------------------------------------
// Stage 2: cur[b,hd,e,l] = (sum_m em[b,e,m] * att[b,hd,mm[m],l]) / (sum_m em + eps)
// ---------------------------------------------------------------------------
__global__ void cur_kernel(const float* __restrict__ att,
                           const float* __restrict__ em,
                           const int*   __restrict__ mm,
                           float* __restrict__ cur) {
  int b = blockIdx.x / HEADS_, hd = blockIdx.x % HEADS_;
  __shared__ float em_s[E_ * M_];
  __shared__ float inv_s[E_];
  __shared__ int   mm_s[M_];
  for (int i = threadIdx.x; i < E_ * M_; i += blockDim.x)
    em_s[i] = em[(long)b * E_ * M_ + i];
  if (threadIdx.x < M_) mm_s[threadIdx.x] = mm[(long)b * M_ + threadIdx.x];
  __syncthreads();
  if (threadIdx.x < E_) {
    float s = 0.f;
    for (int m = 0; m < M_; ++m) s += em_s[threadIdx.x * M_ + m];
    inv_s[threadIdx.x] = 1.f / (s + EPS_);
  }
  __syncthreads();
  const float* att_b = att + ((long)b * HEADS_ + hd) * L_ * L_;
  float* cur_b = cur + ((long)b * HEADS_ + hd) * E_ * L_;
  for (int l = threadIdx.x; l < L_; l += blockDim.x) {
    float acc[E_];
#pragma unroll
    for (int e = 0; e < E_; ++e) acc[e] = 0.f;
    for (int m = 0; m < M_; ++m) {
      float a = att_b[(long)mm_s[m] * L_ + l];
#pragma unroll
      for (int e = 0; e < E_; ++e) acc[e] += em_s[e * M_ + m] * a;
    }
#pragma unroll
    for (int e = 0; e < E_; ++e) cur_b[(long)e * L_ + l] = acc[e] * inv_s[e];
  }
}

// ---------------------------------------------------------------------------
// Stage 3: ca[b,p,l] = sum_hd cur[hd,hi,l]*cur[hd,ti,l], normalized over l (bf16)
// Rows p in [P_, PPAD) are zero padding.
// ---------------------------------------------------------------------------
__global__ void ca_kernel(const float* __restrict__ cur,
                          const int* __restrict__ hts,
                          __bf16* __restrict__ ca) {
  int p = blockIdx.x, b = blockIdx.y;
  __bf16* row = ca + ((long)b * PPAD + p) * L_;
  if (p >= P_) {
    for (int l = threadIdx.x; l < L_; l += 256) row[l] = (__bf16)0.f;
    return;
  }
  int hi = hts[((long)b * 2 + 0) * P_ + p];
  int ti = hts[((long)b * 2 + 1) * P_ + p];
  const float* curb = cur + (long)b * HEADS_ * E_ * L_;
  float v[4];
  float loc = 0.f;
#pragma unroll
  for (int il = 0; il < 4; ++il) {
    int l = threadIdx.x + il * 256;
    float s = 0.f;
#pragma unroll
    for (int hd = 0; hd < HEADS_; ++hd)
      s += curb[((long)hd * E_ + hi) * L_ + l] * curb[((long)hd * E_ + ti) * L_ + l];
    v[il] = s;
    loc += s;
  }
  __shared__ float red[256];
  red[threadIdx.x] = loc;
  __syncthreads();
  for (int st = 128; st > 0; st >>= 1) {
    if (threadIdx.x < st) red[threadIdx.x] += red[threadIdx.x + st];
    __syncthreads();
  }
  float inv = 1.f / (red[0] + EPS_);
#pragma unroll
  for (int il = 0; il < 4; ++il)
    row[threadIdx.x + il * 256] = (__bf16)(v[il] * inv);
}

// ---------------------------------------------------------------------------
// Conversions / gathers
// ---------------------------------------------------------------------------
__global__ void cvt_kernel(const float* __restrict__ src, __bf16* __restrict__ dst,
                           long rows_src, long rows_dst, long cols) {
  long idx = (long)blockIdx.x * blockDim.x + threadIdx.x;
  long total = rows_dst * cols;
  if (idx >= total) return;
  long r = idx / cols;
  dst[idx] = (r < rows_src) ? (__bf16)src[idx] : (__bf16)0.f;
}

// context (B,L,H) f32 -> ctxT (B,H,L) bf16   (N x K layout for the ci GEMM)
__global__ void ctxT_kernel(const float* __restrict__ ctx, __bf16* __restrict__ ctxT) {
  long idx = (long)blockIdx.x * blockDim.x + threadIdx.x;
  long total = (long)B_ * H_ * L_;
  if (idx >= total) return;
  int l = (int)(idx % L_);
  long r = idx / L_;
  int h = (int)(r % H_);
  int b = (int)(r / H_);
  ctxT[idx] = (__bf16)ctx[((long)b * L_ + l) * H_ + h];
}

// gather h/t rows from entity into (PBPAD x H) bf16, row = p*B + b
__global__ void gather_ht_kernel(const float* __restrict__ entity,
                                 const int* __restrict__ hts,
                                 __bf16* __restrict__ hg, __bf16* __restrict__ tg) {
  long idx = (long)blockIdx.x * blockDim.x + threadIdx.x;
  long total = (long)PBPAD * H_;
  if (idx >= total) return;
  int col = (int)(idx % H_);
  long row = idx / H_;
  int p = (int)(row / B_), b = (int)(row % B_);
  if (p < P_) {
    int hi = hts[((long)b * 2 + 0) * P_ + p];
    int ti = hts[((long)b * 2 + 1) * P_ + p];
    hg[idx] = (__bf16)entity[((long)b * E_ + hi) * H_ + col];
    tg[idx] = (__bf16)entity[((long)b * E_ + ti) * H_ + col];
  } else {
    hg[idx] = (__bf16)0.f;
    tg[idx] = (__bf16)0.f;
  }
}

// ---------------------------------------------------------------------------
// GEMM 1: ci[p*B+b, :] = ca[b,p,:] @ context[b]   (880x1024x768, batched over b)
// Each wave: 16 rows x 64 cols (4 accumulators) -> A fragment amortized 4x.
// grid (H/256, PPAD/16, B), 128 threads (4 waves).
// ---------------------------------------------------------------------------
__global__ void gemm_ci_kernel(const __bf16* __restrict__ ca,
                               const __bf16* __restrict__ ctxT,
                               __bf16* __restrict__ ci) {
  int wave = threadIdx.x >> 5;
  int b = blockIdx.z;
  int row0 = blockIdx.y * 16;
  int colbase = blockIdx.x * 256 + wave * 64;
  const __bf16* A = ca + (long)b * PPAD * L_;
  const __bf16* W = ctxT + (long)b * H_ * L_;
  const v8f vz = {0.f, 0.f, 0.f, 0.f, 0.f, 0.f, 0.f, 0.f};
  v8f acc[4] = {vz, vz, vz, vz};
  for (int k = 0; k < L_; k += 32) {
    v16bf a = load_frag_a(A, L_, row0, k);
#pragma unroll
    for (int t = 0; t < 4; ++t)
      acc[t] = wmma_bf16(a, load_frag_b(W, L_, colbase + t * 16, k), acc[t]);
  }
  // output row p maps to global row p*B + b
#pragma unroll
  for (int t = 0; t < 4; ++t)
    store_frag_bf16(ci + (long)b * H_, (long)B_ * H_, row0, colbase + t * 16, acc[t]);
}

// ---------------------------------------------------------------------------
// GEMM 2: out = tanh(A1 @ W1^T + A2 @ W2^T + b1 + b2), M=PBPAD, N=K=768, bf16 out
// Each wave: 16 rows x 64 cols. grid (H/256, PBPAD/16), 128 threads.
// ---------------------------------------------------------------------------
__global__ void gemm_linear_kernel(const __bf16* __restrict__ A1,
                                   const __bf16* __restrict__ W1,
                                   const __bf16* __restrict__ A2,
                                   const __bf16* __restrict__ W2,
                                   const float* __restrict__ b1,
                                   const float* __restrict__ b2,
                                   __bf16* __restrict__ out) {
  int wave = threadIdx.x >> 5, lane = threadIdx.x & 31;
  int row0 = blockIdx.y * 16;
  int colbase = blockIdx.x * 256 + wave * 64;
  const v8f vz = {0.f, 0.f, 0.f, 0.f, 0.f, 0.f, 0.f, 0.f};
  v8f acc[4] = {vz, vz, vz, vz};
  for (int k = 0; k < H_; k += 32) {
    v16bf a = load_frag_a(A1, H_, row0, k);
#pragma unroll
    for (int t = 0; t < 4; ++t)
      acc[t] = wmma_bf16(a, load_frag_b(W1, H_, colbase + t * 16, k), acc[t]);
  }
  for (int k = 0; k < H_; k += 32) {
    v16bf a = load_frag_a(A2, H_, row0, k);
#pragma unroll
    for (int t = 0; t < 4; ++t)
      acc[t] = wmma_bf16(a, load_frag_b(W2, H_, colbase + t * 16, k), acc[t]);
  }
  int rbase = row0 + 8 * (lane >> 4);
#pragma unroll
  for (int t = 0; t < 4; ++t) {
    int col = colbase + t * 16 + (lane & 15);
    float bias = b1[col] + b2[col];
#pragma unroll
    for (int v = 0; v < 8; ++v)
      out[(long)(rbase + v) * H_ + col] = (__bf16)tanhf(acc[t][v] + bias);
  }
}

// ---------------------------------------------------------------------------
// Classifier: out[pb, r] = sum_{n,i,j} h[pb,n,i]*t[pb,n,j]*W[r,n,i,j] + cb[r]
// A fragments of the implicit bl matrix are built on the fly:
//   A(row, K=n*4096+i*64+j) = h[row,n,i] * t[row,n,j]  (scalar x reusable t-frag)
// Block = 256 threads = 8 waves: 2 row-tiles x 4 n-subsets (n = nset + 4*nn).
// Pairs of waves share identical clas_w fragment addresses (WGP$ reuse), and
// 32 rows/block halves total clas_w L2 traffic vs 16 rows. LDS reduce over the
// 4 n-subsets. grid (PBPAD/32).
// ---------------------------------------------------------------------------
__global__ void classifier_kernel(const __bf16* __restrict__ hO,
                                  const __bf16* __restrict__ tO,
                                  const __bf16* __restrict__ cw,
                                  const float* __restrict__ cb,
                                  float* __restrict__ out) {
  int wave = threadIdx.x >> 5, lane = threadIdx.x & 31;
  int rowtile = wave & 1;        // 0..1
  int nset = wave >> 1;          // 0..3
  int row0 = blockIdx.x * 32 + rowtile * 16;
  const v8f vz = {0.f, 0.f, 0.f, 0.f, 0.f, 0.f, 0.f, 0.f};
  v8f acc[7];
#pragma unroll
  for (int t = 0; t < 7; ++t) acc[t] = vz;

  for (int nn = 0; nn < 3; ++nn) {
    int n = nset + nn * 4;
    v16bf tf0 = load_frag_a(tO, H_, row0, (long)n * BLK_);
    v16bf tf1 = load_frag_a(tO, H_, row0, (long)n * BLK_ + 32);
    const __bf16* hrow = hO + (long)(row0 + (lane & 15)) * H_ + n * BLK_;
    for (int i = 0; i < BLK_; ++i) {
      __builtin_prefetch(hrow + i + 16, 0, 3);  // global_prefetch_b8
      float hv = (float)hrow[i];
      v16bf a0 = scale_frag(tf0, hv);
      v16bf a1 = scale_frag(tf1, hv);
      long kb = (long)n * (BLK_ * BLK_) + (long)i * BLK_;
#pragma unroll
      for (int t = 0; t < 7; ++t) {
        acc[t] = wmma_bf16(a0, load_frag_b(cw, KCLS, t * 16, kb), acc[t]);
        acc[t] = wmma_bf16(a1, load_frag_b(cw, KCLS, t * 16, kb + 32), acc[t]);
      }
    }
  }

  __shared__ float lds[8 * 7 * 256];
#pragma unroll
  for (int t = 0; t < 7; ++t)
#pragma unroll
    for (int v = 0; v < 8; ++v)
      lds[((wave * 7 + t) << 8) + lane * 8 + v] = acc[t][v];
  __syncthreads();

  int e = threadIdx.x;           // 0..255: one C element per (rowtile, tile)
  int lane_e = e >> 3, v = e & 7;
  for (int rt = 0; rt < 2; ++rt) {
    int r0 = blockIdx.x * 32 + rt * 16;
    int row = r0 + v + 8 * (lane_e >> 4);
    for (int t = 0; t < 7; ++t) {
      float s = 0.f;
#pragma unroll
      for (int ns = 0; ns < 4; ++ns)
        s += lds[((((ns * 2 + rt) * 7 + t)) << 8) + e];
      int col = t * 16 + (lane_e & 15);
      if (row < P_ * B_ && col < R_)
        out[(long)row * R_ + col] = s + cb[col];
    }
  }
}

// ---------------------------------------------------------------------------
// Host launcher
// ---------------------------------------------------------------------------
extern "C" void kernel_launch(void* const* d_in, const int* in_sizes, int n_in,
                              void* d_out, int out_size, void* d_ws, size_t ws_size,
                              hipStream_t stream) {
  (void)in_sizes; (void)n_in; (void)out_size; (void)ws_size;

  const float* context   = (const float*)d_in[0];
  const float* attention = (const float*)d_in[1];
  const int*   mm        = (const int*)d_in[2];
  const float* em        = (const float*)d_in[3];
  const int*   hts       = (const int*)d_in[4];
  const float* h_w  = (const float*)d_in[5];
  const float* h_b  = (const float*)d_in[6];
  const float* t_w  = (const float*)d_in[7];
  const float* t_b  = (const float*)d_in[8];
  const float* hc_w = (const float*)d_in[9];
  const float* hc_b = (const float*)d_in[10];
  const float* tc_w = (const float*)d_in[11];
  const float* tc_b = (const float*)d_in[12];
  const float* cw   = (const float*)d_in[13];
  const float* cb   = (const float*)d_in[14];
  float* out = (float*)d_out;

  // workspace carve-up (256B aligned)
  char* ws = (char*)d_ws;
  size_t off = 0;
  auto take = [&](size_t bytes) {
    char* p = ws + off;
    off += (bytes + 255) & ~(size_t)255;
    return p;
  };
  float*  entity  = (float*)take((size_t)B_ * E_ * H_ * 4);
  float*  cur     = (float*)take((size_t)B_ * HEADS_ * E_ * L_ * 4);
  __bf16* ca_bf   = (__bf16*)take((size_t)B_ * PPAD * L_ * 2);
  __bf16* ctxT_bf = (__bf16*)take((size_t)B_ * H_ * L_ * 2);
  __bf16* hg_bf   = (__bf16*)take((size_t)PBPAD * H_ * 2);
  __bf16* tg_bf   = (__bf16*)take((size_t)PBPAD * H_ * 2);
  __bf16* ci_bf   = (__bf16*)take((size_t)PBPAD * H_ * 2);
  __bf16* hw_bf   = (__bf16*)take((size_t)H_ * H_ * 2);
  __bf16* tw_bf   = (__bf16*)take((size_t)H_ * H_ * 2);
  __bf16* hcw_bf  = (__bf16*)take((size_t)H_ * H_ * 2);
  __bf16* tcw_bf  = (__bf16*)take((size_t)H_ * H_ * 2);
  __bf16* cw_bf   = (__bf16*)take((size_t)RPAD * KCLS * 2);
  __bf16* hout_bf = (__bf16*)take((size_t)PBPAD * H_ * 2);
  __bf16* tout_bf = (__bf16*)take((size_t)PBPAD * H_ * 2);

  // Stage 1-3: gathers / softmax-like contractions
  entity_kernel<<<B_ * E_, 256, 0, stream>>>(context, em, mm, entity);
  cur_kernel<<<B_ * HEADS_, 256, 0, stream>>>(attention, em, mm, cur);
  ca_kernel<<<dim3(PPAD, B_), 256, 0, stream>>>(cur, hts, ca_bf);

  // bf16 staging
  {
    long tot = (long)B_ * H_ * L_;
    ctxT_kernel<<<(unsigned)((tot + 255) / 256), 256, 0, stream>>>(context, ctxT_bf);
  }
  {
    long tot = (long)H_ * H_;
    unsigned g = (unsigned)((tot + 255) / 256);
    cvt_kernel<<<g, 256, 0, stream>>>(h_w,  hw_bf,  H_, H_, H_);
    cvt_kernel<<<g, 256, 0, stream>>>(t_w,  tw_bf,  H_, H_, H_);
    cvt_kernel<<<g, 256, 0, stream>>>(hc_w, hcw_bf, H_, H_, H_);
    cvt_kernel<<<g, 256, 0, stream>>>(tc_w, tcw_bf, H_, H_, H_);
  }
  {
    long tot = (long)RPAD * KCLS;
    cvt_kernel<<<(unsigned)((tot + 255) / 256), 256, 0, stream>>>(cw, cw_bf, R_, RPAD, KCLS);
  }
  {
    long tot = (long)PBPAD * H_;
    gather_ht_kernel<<<(unsigned)((tot + 255) / 256), 256, 0, stream>>>(entity, hts,
                                                                        hg_bf, tg_bf);
  }

  // WMMA GEMMs
  gemm_ci_kernel<<<dim3(H_ / 256, PPAD / 16, B_), 128, 0, stream>>>(ca_bf, ctxT_bf, ci_bf);
  gemm_linear_kernel<<<dim3(H_ / 256, PBPAD / 16), 128, 0, stream>>>(
      hg_bf, hw_bf, ci_bf, hcw_bf, h_b, hc_b, hout_bf);
  gemm_linear_kernel<<<dim3(H_ / 256, PBPAD / 16), 128, 0, stream>>>(
      tg_bf, tw_bf, ci_bf, tcw_bf, t_b, tc_b, tout_bf);

  // Bilinear classifier (implicit bl construction)
  classifier_kernel<<<PBPAD / 32, 256, 0, stream>>>(hout_bf, tout_bf, cw_bf, cb, out);
}
